// SelfAttention_25555055411689
// MI455X (gfx1250) — matrix operand order repaired and checked by reading, not compiled, not run on hardware
//
#include <hip/hip_runtime.h>
#include <hip/hip_bf16.h>

#define BATCH 4
#define SEQ   4096
#define DIM   1024
#define SCALE 0.03125f   /* 1/sqrt(1024) */

typedef __attribute__((ext_vector_type(16))) __bf16 v16bf;
typedef __attribute__((ext_vector_type(8)))  __bf16 v8bf;
typedef __attribute__((ext_vector_type(4)))  __bf16 v4bf;
typedef __attribute__((ext_vector_type(8)))  float  v8f;

union AFrag { v16bf v; v8bf h[2]; };

// ---------------------------------------------------------------------------
// Kernel 1: fp32 -> bf16 conversion, 4 elements per thread.
// ---------------------------------------------------------------------------
__global__ __launch_bounds__(256) void cvt_f32_bf16(const float* __restrict__ src,
                                                    __bf16* __restrict__ dst, int n4) {
  int i = blockIdx.x * blockDim.x + threadIdx.x;
  if (i < n4) {
    float4 f = ((const float4*)src)[i];
    v4bf o;
    o[0] = (__bf16)f.x; o[1] = (__bf16)f.y; o[2] = (__bf16)f.z; o[3] = (__bf16)f.w;
    ((v4bf*)dst)[i] = o;
  }
}

// ---------------------------------------------------------------------------
// Kernel 2: fused QKV projection, out = x @ W^T (W is [out,in] row-major).
// One wave computes a 16x64 output tile (4 accumulators). The K loop (32
// steps) is FULLY unrolled: all fragments are SSA values, so the scheduler
// hoists load clauses ahead of the consuming WMMAs with no register copies.
// V is stored TRANSPOSED [B, D, S] so the PV B-operand later is contiguous.
// ---------------------------------------------------------------------------
__global__ __launch_bounds__(256) void qkv_gemm(const __bf16* __restrict__ X,
                                                const __bf16* __restrict__ Wq,
                                                const __bf16* __restrict__ Wk,
                                                const __bf16* __restrict__ Wv,
                                                __bf16* __restrict__ Q,
                                                __bf16* __restrict__ K,
                                                __bf16* __restrict__ VT) {
  const int tid  = threadIdx.x;
  const int w    = tid >> 5;
  const int lane = tid & 31;
  const int l16  = lane & 15;
  const int half = lane >> 4;
  const int kb8  = half * 8;
  const int kb16 = half * 16;

  // 1024 m-tiles * 16 n-groups (of 64) * 3 matrices, 8 waves per block
  const unsigned tile = blockIdx.x * 8u + w;
  const int z  = tile >> 14;            // 0=Q 1=K 2=V
  const unsigned rem = tile & 16383u;
  const int m0 = (rem >> 4) << 4;
  const int n0 = (rem & 15u) << 6;

  const __bf16* W = (z == 0) ? Wq : ((z == 1) ? Wk : Wv);
  const __bf16* arow = X + (size_t)(m0 + l16) * DIM + kb16;   // A: row = lane%16
  const __bf16* brow = W + (size_t)(n0 + l16) * DIM + kb16;   // B: col = lane%16

  v8f acc[4];
#pragma unroll
  for (int ai = 0; ai < 4; ++ai) acc[ai] = (v8f){};

#pragma unroll
  for (int k0 = 0; k0 < DIM; k0 += 32) {
    AFrag a;
    a.h[0] = *(const v8bf*)(arow + k0);        // K = kb16..kb16+7
    a.h[1] = *(const v8bf*)(arow + k0 + 8);    // K = kb16+8..kb16+15
    v16bf b[4];
#pragma unroll
    for (int ai = 0; ai < 4; ++ai)
      b[ai] = *(const v16bf*)(brow + (size_t)(ai * 16) * DIM + k0);
#pragma unroll
    for (int ai = 0; ai < 4; ++ai)
      acc[ai] = __builtin_amdgcn_wmma_f32_16x16x32_bf16(false, a.v, false, b[ai],
                                                        (short)0, acc[ai], false, false);
  }

  if (z == 2) {  // V transposed: VT[b][d][s]
#pragma unroll
    for (int ai = 0; ai < 4; ++ai)
#pragma unroll
      for (int r = 0; r < 8; ++r) {
        int m = m0 + r + 8 * half;
        int bI = m >> 12, s = m & (SEQ - 1);
        VT[((size_t)(bI * DIM + n0 + ai * 16 + l16)) * SEQ + s] = (__bf16)acc[ai][r];
      }
  } else {
    __bf16* O = (z == 0) ? Q : K;
#pragma unroll
    for (int ai = 0; ai < 4; ++ai)
#pragma unroll
      for (int r = 0; r < 8; ++r) {
        int m = m0 + r + 8 * half;
        O[(size_t)m * DIM + n0 + ai * 16 + l16] = (__bf16)acc[ai][r];
      }
  }
}

// ---------------------------------------------------------------------------
// Kernel 3: flash attention. One block (8 waves) per 32 query rows (2 row-sets
// of 16). Each wave owns a 128-wide D chunk for both the QK^T partial
// reduction and the PV output accumulation. Key tiles of 32 per iteration;
// every loaded K/V fragment feeds 2 WMMAs (one per row-set).
// ---------------------------------------------------------------------------
__global__ __launch_bounds__(256, 1) void flash_attn(const __bf16* __restrict__ Q,
                                                     const __bf16* __restrict__ K,
                                                     const __bf16* __restrict__ VT,
                                                     float* __restrict__ out) {
  __shared__ float part[32][256];   // [w*4 + u*2 + kt][r*32+lane]  (32 KB)
  __shared__ __attribute__((aligned(16))) __bf16 Plds[2][16 * 32];

  const int tid  = threadIdx.x;
  const int w    = tid >> 5;
  const int lane = tid & 31;
  const int l16  = lane & 15;
  const int half = lane >> 4;
  const int kb8  = half * 8;
  const int kb16 = half * 16;

  const int b  = blockIdx.x >> 7;          // 128 query tiles (of 32) per batch
  const int q0 = (blockIdx.x & 127) << 5;
  const int dchunk = w * 128;

  // Preload this wave's Q A-fragments: 2 row-sets x (16 rows x 128 of D).
  v16bf qf[2][4];
#pragma unroll
  for (int u = 0; u < 2; ++u) {
    const __bf16* qrow = Q + (size_t)(b * SEQ + q0 + u * 16 + l16) * DIM + dchunk;
#pragma unroll
    for (int f = 0; f < 4; ++f) {
      AFrag a;
      a.h[0] = *(const v8bf*)(qrow + f * 32 + kb16);
      a.h[1] = *(const v8bf*)(qrow + f * 32 + kb16 + 8);
      qf[u][f] = a.v;
    }
  }

  v8f oacc[2][8];
#pragma unroll
  for (int u = 0; u < 2; ++u)
#pragma unroll
    for (int t = 0; t < 8; ++t) oacc[u][t] = (v8f){};
  float mrow[2][8], lrow[2][8];
#pragma unroll
  for (int u = 0; u < 2; ++u)
#pragma unroll
    for (int r = 0; r < 8; ++r) { mrow[u][r] = -1e30f; lrow[u][r] = 0.0f; }

  for (int j0 = 0; j0 < SEQ; j0 += 32) {
    // ---- partial scores over this wave's D chunk -------------------------
    const __bf16* kbase = K + (size_t)(b * SEQ + j0 + l16) * DIM + dchunk + kb16;
    if (j0 + 32 < SEQ) __builtin_prefetch(kbase + (size_t)32 * DIM, 0, 1);
    v8f sp[2][2];
#pragma unroll
    for (int u = 0; u < 2; ++u) { sp[u][0] = (v8f){}; sp[u][1] = (v8f){}; }
#pragma unroll
    for (int f = 0; f < 4; ++f) {
      v16bf bk0 = *(const v16bf*)(kbase + f * 32);
      v16bf bk1 = *(const v16bf*)(kbase + (size_t)16 * DIM + f * 32);
#pragma unroll
      for (int u = 0; u < 2; ++u) {
        sp[u][0] = __builtin_amdgcn_wmma_f32_16x16x32_bf16(false, qf[u][f], false, bk0,
                                                           (short)0, sp[u][0], false, false);
        sp[u][1] = __builtin_amdgcn_wmma_f32_16x16x32_bf16(false, qf[u][f], false, bk1,
                                                           (short)0, sp[u][1], false, false);
      }
    }
#pragma unroll
    for (int u = 0; u < 2; ++u)
#pragma unroll
      for (int r = 0; r < 8; ++r) {
        part[w * 4 + u * 2 + 0][r * 32 + lane] = sp[u][0][r];
        part[w * 4 + u * 2 + 1][r * 32 + lane] = sp[u][1][r];
      }
    __syncthreads();

    // ---- reduce partials + online softmax per row-set --------------------
#pragma unroll
    for (int u = 0; u < 2; ++u) {
      float s0[8], s1[8];
#pragma unroll
      for (int r = 0; r < 8; ++r) {
        float a0 = 0.0f, a1 = 0.0f;
#pragma unroll
        for (int ww = 0; ww < 8; ++ww) {
          a0 += part[ww * 4 + u * 2 + 0][r * 32 + lane];
          a1 += part[ww * 4 + u * 2 + 1][r * 32 + lane];
        }
        s0[r] = a0 * SCALE;
        s1[r] = a1 * SCALE;
      }
      float alpha[8];
#pragma unroll
      for (int r = 0; r < 8; ++r) {
        float v = fmaxf(s0[r], s1[r]);
        v = fmaxf(v, __shfl_xor(v, 8, 32));
        v = fmaxf(v, __shfl_xor(v, 4, 32));
        v = fmaxf(v, __shfl_xor(v, 2, 32));
        v = fmaxf(v, __shfl_xor(v, 1, 32));
        float mn = fmaxf(mrow[u][r], v);
        alpha[r] = __expf(mrow[u][r] - mn);
        mrow[u][r] = mn;
        s0[r] = __expf(s0[r] - mn);
        s1[r] = __expf(s1[r] - mn);
        float rs = s0[r] + s1[r];
        rs += __shfl_xor(rs, 8, 32);
        rs += __shfl_xor(rs, 4, 32);
        rs += __shfl_xor(rs, 2, 32);
        rs += __shfl_xor(rs, 1, 32);
        lrow[u][r] = lrow[u][r] * alpha[r] + rs;
      }
#pragma unroll
      for (int t = 0; t < 8; ++t)
#pragma unroll
        for (int r = 0; r < 8; ++r) oacc[u][t][r] *= alpha[r];

      if (w == 0) {   // stage P (16x32 bf16) through LDS: C-frag -> A-frag
#pragma unroll
        for (int r = 0; r < 8; ++r) {
          int m = r + 8 * half;
          Plds[u][m * 32 + l16]      = (__bf16)s0[r];
          Plds[u][m * 32 + 16 + l16] = (__bf16)s1[r];
        }
      }
    }
    __syncthreads();

    AFrag p[2];
#pragma unroll
    for (int u = 0; u < 2; ++u) {
      p[u].h[0] = *(const v8bf*)(&Plds[u][l16 * 32 + kb16]);
      p[u].h[1] = *(const v8bf*)(&Plds[u][l16 * 32 + kb16 + 8]);
    }

    // ---- PV: accumulate both 16x128 output chunks (double-buffered V) ----
    const __bf16* vbase = VT + (size_t)(b * DIM + dchunk + l16) * SEQ + j0 + kb16;
    v16bf bv_cur = *(const v16bf*)vbase;
#pragma unroll
    for (int t = 0; t < 8; ++t) {
      v16bf bv_nxt = bv_cur;
      if (t < 7)
        bv_nxt = *(const v16bf*)(vbase + (size_t)((t + 1) * 16) * SEQ);
#pragma unroll
      for (int u = 0; u < 2; ++u)
        oacc[u][t] = __builtin_amdgcn_wmma_f32_16x16x32_bf16(false, p[u].v, false, bv_cur,
                                                             (short)0, oacc[u][t], false, false);
      bv_cur = bv_nxt;
    }
    // part[]/Plds rewrites of the next iteration are fenced by its barriers.
  }

  // ---- epilogue: normalize and store fp32 --------------------------------
#pragma unroll
  for (int u = 0; u < 2; ++u)
#pragma unroll
    for (int t = 0; t < 8; ++t)
#pragma unroll
      for (int r = 0; r < 8; ++r) {
        int m = r + 8 * half;
        float val = oacc[u][t][r] / lrow[u][r];
        out[(size_t)(b * SEQ + q0 + u * 16 + m) * DIM + dchunk + t * 16 + l16] = val;
      }
}

// ---------------------------------------------------------------------------
extern "C" void kernel_launch(void* const* d_in, const int* in_sizes, int n_in,
                              void* d_out, int out_size, void* d_ws, size_t ws_size,
                              hipStream_t stream) {
  const float* x  = (const float*)d_in[0];   // [4,4096,1024]
  const float* wq = (const float*)d_in[1];   // [1024,1024]
  const float* wk = (const float*)d_in[2];
  const float* wv = (const float*)d_in[3];
  float* out = (float*)d_out;

  const size_t M  = (size_t)BATCH * SEQ;          // 16384
  const size_t NX = M * DIM;                      // 16.7M elements
  const size_t NW = (size_t)DIM * DIM;            // 1.05M elements

  char* ws = (char*)d_ws;
  __bf16* xb  = (__bf16*)ws;                ws += NX * 2;
  __bf16* wqb = (__bf16*)ws;                ws += NW * 2;
  __bf16* wkb = (__bf16*)ws;                ws += NW * 2;
  __bf16* wvb = (__bf16*)ws;                ws += NW * 2;
  __bf16* Qb  = (__bf16*)ws;                ws += NX * 2;
  __bf16* Kb  = (__bf16*)ws;                ws += NX * 2;
  __bf16* VTb = (__bf16*)ws;                /* [B, D, S] */

  // 1) convert to bf16
  cvt_f32_bf16<<<(int)(NX / 4 / 256), 256, 0, stream>>>(x,  xb,  (int)(NX / 4));
  cvt_f32_bf16<<<(int)(NW / 4 / 256), 256, 0, stream>>>(wq, wqb, (int)(NW / 4));
  cvt_f32_bf16<<<(int)(NW / 4 / 256), 256, 0, stream>>>(wk, wkb, (int)(NW / 4));
  cvt_f32_bf16<<<(int)(NW / 4 / 256), 256, 0, stream>>>(wv, wvb, (int)(NW / 4));

  // 2) QKV projections: 1024 m-tiles * 16 n-groups * 3, 8 waves per block
  qkv_gemm<<<(1024 * 16 * 3) / 8, 256, 0, stream>>>(xb, wqb, wkb, wvb, Qb, Kb, VTb);

  // 3) flash attention: one block per 32 query rows
  flash_attn<<<BATCH * SEQ / 32, 256, 0, stream>>>(Qb, Kb, VTb, out);
}
